// PolynomialKernel_81140522156463
// MI455X (gfx1250) — matrix operand order repaired and checked by reading, not compiled, not run on hardware
//
#include <hip/hip_runtime.h>

// ---------------------------------------------------------------------------
// Batched Gram + polynomial epilogue for MI455X (gfx1250, wave32, WMMA).
//   out[n,c,k] = (sum_m x[n,c,m]*x[n,k,m] + 1)^2
// x: (512, 128, 512) fp32.  One workgroup per batch.  fp32 is split into
// bf16 hi/lo and the product computed as hi*hi + hi*lo + lo*hi with
// v_wmma_f32_16x16x32_bf16 (~fp32 accuracy, 8x the K-depth of fp32 WMMA).
// Double-buffered LDS: next K-chunk is staged in registers (non-temporal
// global_load_b128) while the current chunk feeds the matrix pipes; one
// workgroup barrier per chunk.
// ---------------------------------------------------------------------------

typedef __attribute__((ext_vector_type(16))) __bf16    v16bf;
typedef __attribute__((ext_vector_type(8)))  float     v8f;
typedef __attribute__((ext_vector_type(4)))  float     f32x4;
typedef __attribute__((ext_vector_type(4)))  unsigned  u32x4;
typedef __attribute__((ext_vector_type(2)))  unsigned  u32x2;

#define C_DIM 128            // rows of X per batch (output is C_DIM x C_DIM)
#define M_DIM 512            // reduction length
#define KC    64             // K-chunk staged in LDS
#define NCH   (M_DIM / KC)   // 8 chunks
#define LSTR  72             // LDS row stride in bf16 elems: 144B = 36 dwords
                             // (odd multiple of 4 dwords -> bank-friendly,
                             //  16B-aligned rows for ds_load_b128)

union FragCast { u32x4 u[2]; v16bf v; };

__device__ __forceinline__ unsigned pack_hi(float a, float b) {
  // truncate-to-bf16 of two floats, packed into one dword
  return (__float_as_uint(a) >> 16) | (__float_as_uint(b) & 0xffff0000u);
}

__global__ __launch_bounds__(256) void gram_poly_wmma(
    const float* __restrict__ X, float* __restrict__ Out) {
  // double-buffered split-bf16 panels: 2 * 2 * 128*72*2B = 72 KB
  __shared__ unsigned short lds_hi[2][C_DIM * LSTR];
  __shared__ unsigned short lds_lo[2][C_DIM * LSTR];

  const int n    = blockIdx.x;
  const int tid  = threadIdx.x;
  const int wave = tid >> 5;          // 8 waves: wave w owns rows 16w..16w+15
  const int lane = tid & 31;
  const int lh   = (lane >> 4) & 1;   // lane half selects K sub-range
  const int ln   = lane & 15;

  const float* Xn = X + (size_t)n * (C_DIM * M_DIM);

  v8f acc[8];
#pragma unroll
  for (int j = 0; j < 8; ++j) acc[j] = (v8f){0.f,0.f,0.f,0.f,0.f,0.f,0.f,0.f};

  f32x4 regs[8];

  auto load_chunk = [&](int kc0) {
#pragma unroll
    for (int i = 0; i < 8; ++i) {
      const int j = tid + 256 * i;    // float4 index, 2048 per panel
      const int c = j >> 4;           // 16 float4 per row
      const int q = j & 15;
      regs[i] = __builtin_nontemporal_load(
          (const f32x4*)(Xn + c * M_DIM + kc0 + 4 * q));
    }
  };

  auto store_chunk = [&](int buf) {
#pragma unroll
    for (int i = 0; i < 8; ++i) {
      const int j = tid + 256 * i;
      const int c = j >> 4;
      const int q = j & 15;
      const f32x4 v = regs[i];

      const unsigned hx = __float_as_uint(v[0]) & 0xffff0000u;
      const unsigned hy = __float_as_uint(v[1]) & 0xffff0000u;
      const unsigned hz = __float_as_uint(v[2]) & 0xffff0000u;
      const unsigned hw = __float_as_uint(v[3]) & 0xffff0000u;
      const float lx = v[0] - __uint_as_float(hx);
      const float ly = v[1] - __uint_as_float(hy);
      const float lz = v[2] - __uint_as_float(hz);
      const float lw = v[3] - __uint_as_float(hw);

      const unsigned off = c * LSTR + 4 * q;        // bf16-element offset
      *(u32x2*)(lds_hi[buf] + off) =
          (u32x2){ (hx >> 16) | hy, (hz >> 16) | hw };
      *(u32x2*)(lds_lo[buf] + off) =
          (u32x2){ pack_hi(lx, ly), pack_hi(lz, lw) };
    }
  };

  auto compute_chunk = [&](int buf) {
    const unsigned short* arow_hi = lds_hi[buf] + (16 * wave + ln) * LSTR;
    const unsigned short* arow_lo = lds_lo[buf] + (16 * wave + ln) * LSTR;
#pragma unroll
    for (int ko = 0; ko < KC; ko += 32) {
      // A fragment (16x32 bf16): VGPR0-3 K = ko + 8*lh + 0..7,
      //                          VGPR4-7 K = ko + 16 + 8*lh + 0..7
      FragCast ahi, alo;
      ahi.u[0] = *(const u32x4*)(arow_hi + ko + 8 * lh);
      ahi.u[1] = *(const u32x4*)(arow_hi + ko + 16 + 8 * lh);
      alo.u[0] = *(const u32x4*)(arow_lo + ko + 8 * lh);
      alo.u[1] = *(const u32x4*)(arow_lo + ko + 16 + 8 * lh);
#pragma unroll
      for (int bj = 0; bj < 8; ++bj) {
        // B fragment (32x16 bf16): lane = column, K = 8*lh + 0..7, 16 + 8*lh + 0..7
        const unsigned short* brow_hi =
            lds_hi[buf] + (16 * bj + ln) * LSTR + ko + 8 * lh;
        const unsigned short* brow_lo =
            lds_lo[buf] + (16 * bj + ln) * LSTR + ko + 8 * lh;
        FragCast bhi, blo;
        bhi.u[0] = *(const u32x4*)(brow_hi);
        bhi.u[1] = *(const u32x4*)(brow_hi + 16);
        blo.u[0] = *(const u32x4*)(brow_lo);
        blo.u[1] = *(const u32x4*)(brow_lo + 16);

        // 3-term compensated product: hi*hi + hi*lo + lo*hi
        acc[bj] = __builtin_amdgcn_wmma_f32_16x16x32_bf16(
            false, ahi.v, false, bhi.v, (short)0, acc[bj], false, false);
        acc[bj] = __builtin_amdgcn_wmma_f32_16x16x32_bf16(
            false, ahi.v, false, blo.v, (short)0, acc[bj], false, false);
        acc[bj] = __builtin_amdgcn_wmma_f32_16x16x32_bf16(
            false, alo.v, false, bhi.v, (short)0, acc[bj], false, false);
      }
    }
  };

  // ---- pipelined main loop ----------------------------------------------
  load_chunk(0);
  store_chunk(0);
  __syncthreads();

  for (int chunk = 0; chunk < NCH; ++chunk) {
    const int cur = chunk & 1;
    if (chunk + 1 < NCH) load_chunk((chunk + 1) * KC);   // prefetch in flight
    compute_chunk(cur);                                  // 48 WMMAs
    if (chunk + 1 < NCH) {
      store_chunk(cur ^ 1);    // safe: all waves finished reading buf^1 before
      __syncthreads();         // the previous barrier; stores precede this one
    }
  }

  // ---- fused epilogue (P + 1)^2, non-temporal stores --------------------
  float* outp = Out + (size_t)n * (C_DIM * C_DIM);
#pragma unroll
  for (int bj = 0; bj < 8; ++bj) {
#pragma unroll
    for (int r = 0; r < 8; ++r) {
      const int c = 16 * wave + 8 * lh + r;   // C/D layout: M = 8*half + r
      const int k = 16 * bj + ln;             // N = lane & 15
      const float p = acc[bj][r] + 1.0f;
      __builtin_nontemporal_store(p * p, outp + c * C_DIM + k);
    }
  }
}

extern "C" void kernel_launch(void* const* d_in, const int* in_sizes, int n_in,
                              void* d_out, int out_size, void* d_ws,
                              size_t ws_size, hipStream_t stream) {
  (void)n_in; (void)out_size; (void)d_ws; (void)ws_size;
  const float* X = (const float*)d_in[0];
  float* O = (float*)d_out;
  const int N = in_sizes[0] / (C_DIM * M_DIM);   // 512 batches
  gram_poly_wmma<<<dim3(N), dim3(256), 0, stream>>>(X, O);
}
